// MoE_32066225832175
// MI455X (gfx1250) — compile-verified, matrix operand order checked by reading, not run on hardware
//
#include <hip/hip_runtime.h>
#include <hip/hip_bf16.h>

// ---------------- problem constants (match reference) ----------------
#define DIM      1024
#define HID      2048
#define NUM_E    8
#define TOPK     2
#define TOKENS   (8 * 2048)          // B*N = 16384
#define ASSIGN   (TOKENS * TOPK)     // 32768

// ---------------- tile config ----------------
#define BM 128
#define BN 128
#define BK 32
#define NTHREADS 256
#define ITERS1 (DIM / BK)            // 32
#define ITERS2 (HID / BK)            // 64

typedef __bf16 bf16;
typedef __attribute__((ext_vector_type(16))) __bf16 v16bf;
typedef __attribute__((ext_vector_type(8)))  __bf16 v8bf;
typedef __attribute__((ext_vector_type(8)))  float  v8f;
typedef int v4i __attribute__((vector_size(16)));
typedef __attribute__((address_space(1))) v4i gv4i;   // global-AS 16B vector
typedef __attribute__((address_space(3))) v4i lv4i;   // LDS-AS 16B vector

// ---------------- workspace layout (bytes) ----------------
#define W1T_OFF   0ull                                    // bf16 [E][HID][DIM]
#define W2T_OFF   (W1T_OFF + (size_t)NUM_E*HID*DIM*2)     // bf16 [E][DIM][HID]
#define H_OFF     (W2T_OFF + (size_t)NUM_E*DIM*HID*2)     // bf16 [ASSIGN][HID]
#define TOK_OFF   (H_OFF   + (size_t)ASSIGN*HID*2)        // int  [ASSIGN]
#define SC_OFF    (TOK_OFF + (size_t)ASSIGN*4)            // f32  [ASSIGN]
#define TIDX_OFF  (SC_OFF  + (size_t)ASSIGN*4)            // int  [TOKENS][2]
#define TSC_OFF   (TIDX_OFF+ (size_t)TOKENS*2*4)          // f32  [TOKENS][2]
#define CTRL_OFF  (TSC_OFF + (size_t)TOKENS*2*4)          // counts[8], cursors[8], offs[8]

// ---------------- CDNA5 async global->LDS copy ----------------
#if defined(__gfx1250__) && __has_builtin(__builtin_amdgcn_global_load_async_to_lds_b128)
#define HAVE_ASYNC_LDS 1
#else
#define HAVE_ASYNC_LDS 0
#endif

__device__ __forceinline__ void g2l_b128(void* lds_dst, const void* gsrc) {
#if HAVE_ASYNC_LDS
    __builtin_amdgcn_global_load_async_to_lds_b128(
        (gv4i*)gsrc, (lv4i*)lds_dst, 0, 0);
#else
    *(v8bf*)lds_dst = *(const v8bf*)gsrc;
#endif
}

__device__ __forceinline__ void async_wait0() {
#if HAVE_ASYNC_LDS
#if __has_builtin(__builtin_amdgcn_s_wait_asynccnt)
    __builtin_amdgcn_s_wait_asynccnt(0);
#else
    asm volatile("s_wait_asynccnt 0" ::: "memory");
#endif
#endif
}

// ---------------- helpers ----------------
__device__ __forceinline__ v16bf ld_frag(const bf16* row, int koff) {
    // 16-bit operand layout: lanes 0-15 hold K {0..7,16..23}, lanes 16-31 hold K {8..15,24..31}
    v8bf lo = *(const v8bf*)(row + koff);
    v8bf hi = *(const v8bf*)(row + 16 + koff);
    return __builtin_shufflevector(lo, hi, 0,1,2,3,4,5,6,7,8,9,10,11,12,13,14,15);
}

__device__ __forceinline__ float silu_f(float v) {
    return v * __builtin_amdgcn_rcpf(1.0f + __expf(-v));
}

// ---------------- kernel 1: fp32 -> bf16 convert + transpose ----------------
// src: [E][K][N] fp32 row-major, dst: [E][N][K] bf16.
// Thread id layout: n fastest -> the 8 strided reads are wave-coalesced;
// each thread writes one contiguous 16B bf16 chunk (8 k values).
__global__ __launch_bounds__(256) void moe_convert_tr_kernel(
    const float* __restrict__ src, bf16* __restrict__ dst, int K, int N) {
    long id = (long)blockIdx.x * blockDim.x + threadIdx.x;
    long total = (long)NUM_E * N * (K / 8);
    if (id >= total) return;
    int n = (int)(id % N);
    long r = id / N;                      // e*(K/8) + kblk
    int kblk = (int)(r % (K / 8));
    int e = (int)(r / (K / 8));
    const float* s = src + (long)e * K * N + (long)kblk * 8 * N + n;
    v8bf p;
#pragma unroll
    for (int j = 0; j < 8; ++j) p[j] = (bf16)s[(long)j * N];
    *(v8bf*)(dst + ((long)e * N + n) * K + kblk * 8) = p;
}

// ---------------- kernel 2: gating (wave per token) ----------------
__global__ __launch_bounds__(256) void moe_gate_kernel(
    const float* __restrict__ x, const float* __restrict__ gw,
    const float* __restrict__ gb, int* __restrict__ tidx,
    float* __restrict__ tsc, int* __restrict__ counts) {
    const int wid = threadIdx.x >> 5, lane = threadIdx.x & 31;
    const int t = blockIdx.x * 8 + wid;   // 8 waves per block, 1 token per wave
    if (t >= TOKENS) return;
    float acc[NUM_E];
#pragma unroll
    for (int e = 0; e < NUM_E; ++e) acc[e] = 0.0f;
    const float* xr = x + (long)t * DIM;
    for (int d = lane; d < DIM; d += 32) {      // coalesced across lanes
        float xv = xr[d];
        const float* gr = gw + d * NUM_E;
#pragma unroll
        for (int e = 0; e < NUM_E; ++e) acc[e] = fmaf(xv, gr[e], acc[e]);
    }
#pragma unroll
    for (int off = 16; off > 0; off >>= 1)
#pragma unroll
        for (int e = 0; e < NUM_E; ++e) acc[e] += __shfl_xor(acc[e], off, 32);
    if (lane == 0) {
#pragma unroll
        for (int e = 0; e < NUM_E; ++e) acc[e] += gb[e];
        int i0 = 0; float v0 = acc[0];
#pragma unroll
        for (int e = 1; e < NUM_E; ++e) if (acc[e] > v0) { v0 = acc[e]; i0 = e; }
        int i1 = -1; float v1 = -3.4e38f;
#pragma unroll
        for (int e = 0; e < NUM_E; ++e) if (e != i0 && acc[e] > v1) { v1 = acc[e]; i1 = e; }
        float e1 = __expf(v1 - v0);
        float inv = __builtin_amdgcn_rcpf(1.0f + e1);
        tidx[2 * t + 0] = i0;  tidx[2 * t + 1] = i1;
        tsc [2 * t + 0] = inv; tsc [2 * t + 1] = e1 * inv;
        atomicAdd(&counts[i0], 1);
        atomicAdd(&counts[i1], 1);
    }
}

// ---------------- kernel 3: 8-entry exclusive scan ----------------
__global__ void moe_offsets_kernel(const int* __restrict__ counts,
                                   int* __restrict__ offs) {
    if (threadIdx.x == 0) {
        int s = 0;
        for (int e = 0; e < NUM_E; ++e) { offs[e] = s; s += counts[e]; }
    }
}

// ---------------- kernel 4: scatter into compacted expert buckets ----------
__global__ __launch_bounds__(256) void moe_scatter_kernel(
    const int* __restrict__ tidx, const float* __restrict__ tsc,
    const int* __restrict__ offs, int* __restrict__ cursors,
    int* __restrict__ tok_list, float* __restrict__ score_list) {
    int t = blockIdx.x * blockDim.x + threadIdx.x;
    if (t >= TOKENS) return;
#pragma unroll
    for (int k = 0; k < TOPK; ++k) {
        int e = tidx[2 * t + k];
        int pos = atomicAdd(&cursors[e], 1);
        int a = offs[e] + pos;
        tok_list[a] = t;
        score_list[a] = tsc[2 * t + k];
    }
}

// ---------------- kernel 5: grouped GEMM1  h = silu(x @ w1 + b1) -----------
// grid: (HID/BN, TOKENS/BM, NUM_E), block 256 (8 waves, wave32)
// Double-buffered LDS; B staged async global->LDS; A gathered fp32 -> bf16.
__global__ __launch_bounds__(NTHREADS) void moe_gemm1_silu_kernel(
    const float* __restrict__ x, const bf16* __restrict__ w1t,
    const float* __restrict__ b1, const int* __restrict__ tok_list,
    const int* __restrict__ counts, const int* __restrict__ offs,
    bf16* __restrict__ h) {
    const int e = blockIdx.z, mt = blockIdx.y, nt = blockIdx.x;
    const int cnt = counts[e];
    if (mt * BM >= cnt) return;               // uniform early exit
    const int base = offs[e];

    __shared__ __align__(16) bf16 As[2][BM][BK];
    __shared__ __align__(16) bf16 Bs[2][BN][BK];

    const int tid = threadIdx.x;
    const int r = tid >> 1;                    // staging row 0..127
    const int halfk = (tid & 1) * 16;          // k sub-offset 0 / 16

    const int gmr = mt * BM + r;
    const int slot_r = base + (gmr < cnt ? gmr : cnt - 1);
    const float* xrow = x + (long)tok_list[slot_r] * DIM + halfk;
    const bf16*  brow = w1t + ((long)e * HID + (long)nt * BN + r) * DIM + halfk;

    const int wid = tid >> 5, lane = tid & 31;
    const int wm = (wid & 3) * 32;             // 4 waves x 32 rows
    const int wn = (wid >> 2) * 64;            // 2 waves x 64 cols
    const int lr = lane & 15;
    const int koff = (lane >> 4) * 8;

    v8f c[2][4];
#pragma unroll
    for (int im = 0; im < 2; ++im)
#pragma unroll
        for (int in = 0; in < 4; ++in)
#pragma unroll
            for (int q = 0; q < 8; ++q) c[im][in][q] = 0.0f;

    float4 f0, f1, f2, f3;
    // ---- prologue: tile 0 ----
    {
        const float4* xp = (const float4*)xrow;
        f0 = xp[0]; f1 = xp[1]; f2 = xp[2]; f3 = xp[3];
        g2l_b128(&Bs[0][r][halfk],     brow);
        g2l_b128(&Bs[0][r][halfk + 8], brow + 8);
        v8bf p0, p1;
        p0[0]=(bf16)f0.x; p0[1]=(bf16)f0.y; p0[2]=(bf16)f0.z; p0[3]=(bf16)f0.w;
        p0[4]=(bf16)f1.x; p0[5]=(bf16)f1.y; p0[6]=(bf16)f1.z; p0[7]=(bf16)f1.w;
        p1[0]=(bf16)f2.x; p1[1]=(bf16)f2.y; p1[2]=(bf16)f2.z; p1[3]=(bf16)f2.w;
        p1[4]=(bf16)f3.x; p1[5]=(bf16)f3.y; p1[6]=(bf16)f3.z; p1[7]=(bf16)f3.w;
        *(v8bf*)&As[0][r][halfk]     = p0;
        *(v8bf*)&As[0][r][halfk + 8] = p1;
    }

    for (int i = 0; i < ITERS1; ++i) {
        const int cur = i & 1, nx = cur ^ 1;
        async_wait0();
        __syncthreads();
        const bool more = (i + 1 < ITERS1);
        if (more) {
            // issue next-tile global traffic before compute (latency hiding)
            const float4* xp = (const float4*)(xrow + (i + 1) * BK);
            f0 = xp[0]; f1 = xp[1]; f2 = xp[2]; f3 = xp[3];
            const bf16* bsrc = brow + (i + 1) * BK;
            g2l_b128(&Bs[nx][r][halfk],     bsrc);
            g2l_b128(&Bs[nx][r][halfk + 8], bsrc + 8);
        }
        v16bf af[2], bfr[4];
#pragma unroll
        for (int im = 0; im < 2; ++im) af[im] = ld_frag(&As[cur][wm + im * 16 + lr][0], koff);
#pragma unroll
        for (int in = 0; in < 4; ++in) bfr[in] = ld_frag(&Bs[cur][wn + in * 16 + lr][0], koff);
#pragma unroll
        for (int im = 0; im < 2; ++im)
#pragma unroll
            for (int in = 0; in < 4; ++in)
                c[im][in] = __builtin_amdgcn_wmma_f32_16x16x32_bf16(
                    false, af[im], false, bfr[in], (short)0, c[im][in], false, false);
        if (more) {
            v8bf p0, p1;
            p0[0]=(bf16)f0.x; p0[1]=(bf16)f0.y; p0[2]=(bf16)f0.z; p0[3]=(bf16)f0.w;
            p0[4]=(bf16)f1.x; p0[5]=(bf16)f1.y; p0[6]=(bf16)f1.z; p0[7]=(bf16)f1.w;
            p1[0]=(bf16)f2.x; p1[1]=(bf16)f2.y; p1[2]=(bf16)f2.z; p1[3]=(bf16)f2.w;
            p1[4]=(bf16)f3.x; p1[5]=(bf16)f3.y; p1[6]=(bf16)f3.z; p1[7]=(bf16)f3.w;
            *(v8bf*)&As[nx][r][halfk]     = p0;
            *(v8bf*)&As[nx][r][halfk + 8] = p1;
        }
    }

    // epilogue: bias + silu, store bf16 h rows
    float b1v[4];
#pragma unroll
    for (int in = 0; in < 4; ++in) b1v[in] = b1[e * HID + nt * BN + wn + in * 16 + lr];
#pragma unroll
    for (int im = 0; im < 2; ++im)
#pragma unroll
        for (int q = 0; q < 8; ++q) {
            int mloc = wm + im * 16 + q + ((lane >> 4) << 3);
            int gm = mt * BM + mloc;
            if (gm < cnt) {
                bf16* ho = h + (long)(base + gm) * HID + nt * BN;
#pragma unroll
                for (int in = 0; in < 4; ++in) {
                    float v = c[im][in][q] + b1v[in];
                    ho[wn + in * 16 + lr] = (bf16)silu_f(v);
                }
            }
        }
}

// ---------------- kernel 6: grouped GEMM2 + scaled atomic combine ----------
// grid: (DIM/BN, TOKENS/BM, NUM_E), block 256. Both tiles staged async.
__global__ __launch_bounds__(NTHREADS) void moe_gemm2_combine_kernel(
    const bf16* __restrict__ h, const bf16* __restrict__ w2t,
    const float* __restrict__ b2, const int* __restrict__ tok_list,
    const float* __restrict__ score_list, const int* __restrict__ counts,
    const int* __restrict__ offs, float* __restrict__ out) {
    const int e = blockIdx.z, mt = blockIdx.y, nt = blockIdx.x;
    const int cnt = counts[e];
    if (mt * BM >= cnt) return;
    const int base = offs[e];

    __shared__ __align__(16) bf16 As[2][BM][BK];
    __shared__ __align__(16) bf16 Bs[2][BN][BK];

    const int tid = threadIdx.x;
    const int r = tid >> 1;
    const int halfk = (tid & 1) * 16;

    const int gmr = mt * BM + r;
    const int slot_r = base + (gmr < cnt ? gmr : cnt - 1);
    const bf16* arow = h + (long)slot_r * HID + halfk;
    const bf16* brow = w2t + ((long)e * DIM + (long)nt * BN + r) * HID + halfk;

    const int wid = tid >> 5, lane = tid & 31;
    const int wm = (wid & 3) * 32;
    const int wn = (wid >> 2) * 64;
    const int lr = lane & 15;
    const int koff = (lane >> 4) * 8;

    v8f c[2][4];
#pragma unroll
    for (int im = 0; im < 2; ++im)
#pragma unroll
        for (int in = 0; in < 4; ++in)
#pragma unroll
            for (int q = 0; q < 8; ++q) c[im][in][q] = 0.0f;

    // ---- prologue: tile 0 ----
    g2l_b128(&As[0][r][halfk],     arow);
    g2l_b128(&As[0][r][halfk + 8], arow + 8);
    g2l_b128(&Bs[0][r][halfk],     brow);
    g2l_b128(&Bs[0][r][halfk + 8], brow + 8);

    for (int i = 0; i < ITERS2; ++i) {
        const int cur = i & 1, nx = cur ^ 1;
        async_wait0();
        __syncthreads();
        if (i + 1 < ITERS2) {
            const bf16* as = arow + (i + 1) * BK;
            const bf16* bs = brow + (i + 1) * BK;
            g2l_b128(&As[nx][r][halfk],     as);
            g2l_b128(&As[nx][r][halfk + 8], as + 8);
            g2l_b128(&Bs[nx][r][halfk],     bs);
            g2l_b128(&Bs[nx][r][halfk + 8], bs + 8);
        }
        v16bf af[2], bfr[4];
#pragma unroll
        for (int im = 0; im < 2; ++im) af[im] = ld_frag(&As[cur][wm + im * 16 + lr][0], koff);
#pragma unroll
        for (int in = 0; in < 4; ++in) bfr[in] = ld_frag(&Bs[cur][wn + in * 16 + lr][0], koff);
#pragma unroll
        for (int im = 0; im < 2; ++im)
#pragma unroll
            for (int in = 0; in < 4; ++in)
                c[im][in] = __builtin_amdgcn_wmma_f32_16x16x32_bf16(
                    false, af[im], false, bfr[in], (short)0, c[im][in], false, false);
    }

    // epilogue: out[tok] += score * (acc + b2)  (exactly 2 contributions/token)
    float b2v[4];
#pragma unroll
    for (int in = 0; in < 4; ++in) b2v[in] = b2[e * DIM + nt * BN + wn + in * 16 + lr];
#pragma unroll
    for (int im = 0; im < 2; ++im)
#pragma unroll
        for (int q = 0; q < 8; ++q) {
            int mloc = wm + im * 16 + q + ((lane >> 4) << 3);
            int gm = mt * BM + mloc;
            if (gm < cnt) {
                int slot = base + gm;
                int tok = tok_list[slot];
                float sc = score_list[slot];
                float* orow = out + (long)tok * DIM + nt * BN;
#pragma unroll
                for (int in = 0; in < 4; ++in)
                    unsafeAtomicAdd(&orow[wn + in * 16 + lr], sc * (c[im][in][q] + b2v[in]));
            }
        }
}

// ---------------- host-side launch ----------------
extern "C" void kernel_launch(void* const* d_in, const int* in_sizes, int n_in,
                              void* d_out, int out_size, void* d_ws, size_t ws_size,
                              hipStream_t stream) {
    const float* x      = (const float*)d_in[0];
    const float* gate_w = (const float*)d_in[1];
    const float* gate_b = (const float*)d_in[2];
    const float* w1     = (const float*)d_in[3];
    const float* b1     = (const float*)d_in[4];
    const float* w2     = (const float*)d_in[5];
    const float* b2     = (const float*)d_in[6];
    float* out = (float*)d_out;

    char* ws = (char*)d_ws;
    bf16*  w1t        = (bf16*)(ws + W1T_OFF);
    bf16*  w2t        = (bf16*)(ws + W2T_OFF);
    bf16*  h          = (bf16*)(ws + H_OFF);
    int*   tok_list   = (int*)(ws + TOK_OFF);
    float* score_list = (float*)(ws + SC_OFF);
    int*   tidx       = (int*)(ws + TIDX_OFF);
    float* tsc        = (float*)(ws + TSC_OFF);
    int*   counts     = (int*)(ws + CTRL_OFF);
    int*   cursors    = counts + 8;
    int*   offs       = counts + 16;

    (void)hipMemsetAsync(counts, 0, 64, stream);
    (void)hipMemsetAsync(out, 0, (size_t)TOKENS * DIM * sizeof(float), stream);

    // 1) weight conversion (+transpose to [E][N][K] bf16)
    {
        long n1 = (long)NUM_E * HID * (DIM / 8);
        moe_convert_tr_kernel<<<(unsigned)((n1 + 255) / 256), 256, 0, stream>>>(w1, w1t, DIM, HID);
        long n2 = (long)NUM_E * DIM * (HID / 8);
        moe_convert_tr_kernel<<<(unsigned)((n2 + 255) / 256), 256, 0, stream>>>(w2, w2t, HID, DIM);
    }

    // 2) gating (wave per token)
    moe_gate_kernel<<<TOKENS / 8, 256, 0, stream>>>(x, gate_w, gate_b, tidx, tsc, counts);

    // 3) offsets
    moe_offsets_kernel<<<1, 32, 0, stream>>>(counts, offs);

    // 4) scatter
    moe_scatter_kernel<<<(TOKENS + 255) / 256, 256, 0, stream>>>(tidx, tsc, offs, cursors,
                                                                 tok_list, score_list);

    // 5) grouped GEMM1 + silu
    {
        dim3 g(HID / BN, TOKENS / BM, NUM_E);
        moe_gemm1_silu_kernel<<<g, NTHREADS, 0, stream>>>(x, w1t, b1, tok_list, counts, offs, h);
    }

    // 6) grouped GEMM2 + weighted combine
    {
        dim3 g(DIM / BN, TOKENS / BM, NUM_E);
        moe_gemm2_combine_kernel<<<g, NTHREADS, 0, stream>>>(h, w2t, b2, tok_list, score_list,
                                                             counts, offs, out);
    }
}
